// FFDense_7352984011100
// MI455X (gfx1250) — matrix operand order repaired and checked by reading, not compiled, not run on hardware
//
#include <hip/hip_runtime.h>
#include <math.h>

// ---------------------------------------------------------------------------
// FFDense for MI455X (gfx1250): relu(normalize_rows(x) @ W + b)
// bf16 WMMA (v_wmma_f32_16x16x32_bf16), normalization fused into A staging.
// Double-buffered LDS; waves tiled 4(M)x2(N) so each wave does 8 WMMAs per
// 12 ds_load_b128; sched_barrier splits frag-load phase from WMMA chain so a
// single s_wait_dscnt gates 8 back-to-back WMMAs.
// ---------------------------------------------------------------------------

typedef __attribute__((ext_vector_type(16))) __bf16 v16bf;
typedef __attribute__((ext_vector_type(8)))  __bf16 v8bf;
typedef __attribute__((ext_vector_type(4)))  __bf16 v4bf;
typedef __attribute__((ext_vector_type(8)))  float  v8f;

#define NROWS 16384
#define D_IN  4096
#define UNITS 4096
#define LDS_STRIDE 40   // halves per LDS row (32 data + 8 pad) -> 80 bytes

// --- Kernel 1: per-row inverse L2 norm (one wave32 per row) ----------------
__global__ __launch_bounds__(256) void rownorm_kernel(const float* __restrict__ x,
                                                      float* __restrict__ inv_norm) {
  const int row  = blockIdx.x * 8 + (threadIdx.x >> 5);
  const int lane = threadIdx.x & 31;
  const float4* xr = (const float4*)(x + (size_t)row * D_IN);
  float s = 0.f;
#pragma unroll 4
  for (int i = lane; i < D_IN / 4; i += 32) {
    float4 v = xr[i];
    s += v.x * v.x + v.y * v.y + v.z * v.z + v.w * v.w;
  }
#pragma unroll
  for (int off = 16; off > 0; off >>= 1) s += __shfl_xor(s, off, 32);
  if (lane == 0) inv_norm[row] = 1.0f / (sqrtf(s) + 1e-5f);
}

// --- Kernel 2: tiled bf16 WMMA GEMM + bias + ReLU --------------------------
// Block: 256 threads (8 waves). Block tile: 128(M) x 128(N).
// Wave grid 4(M) x 2(N): wave covers 32 rows (2 tiles) x 64 cols (4 tiles).
__global__ __launch_bounds__(256) void ffdense_wmma_kernel(
    const float* __restrict__ x, const float* __restrict__ W,
    const float* __restrict__ bias, const float* __restrict__ inv_norm,
    float* __restrict__ out) {
  __shared__ __align__(16) __bf16 Alds[2][128 * LDS_STRIDE];  // [m][k], k-contig
  __shared__ __align__(16) __bf16 Blds[2][128 * LDS_STRIDE];  // [n][k], k-contig

  const int tid   = threadIdx.x;
  const int wave  = tid >> 5;
  const int lane  = tid & 31;
  const int m_blk = blockIdx.y * 128;
  const int n_blk = blockIdx.x * 128;

  const int wm = wave & 3;   // wave M index: rows wm*32 .. +32
  const int wn = wave >> 2;  // wave N index: cols wn*64 .. +64

  // --- A staging plan: 4 float4 loads/thread cover the 128x32 fp32 tile ---
  const float* xptr[4];
  float scale[4];
  int arow[4], akv[4];
#pragma unroll
  for (int i = 0; i < 4; ++i) {
    int idx = tid + i * 256;          // 0..1023
    arow[i] = idx >> 3;               // 0..127
    akv[i]  = idx & 7;                // float4 slot within 32-wide k
    xptr[i] = x + (size_t)(m_blk + arow[i]) * D_IN + akv[i] * 4;
    scale[i] = inv_norm[m_blk + arow[i]];
  }
  // --- B staging plan: thread owns column n_blk+nloc, 16 k values ---------
  const int nloc = tid & 127;
  const int bk   = (tid >> 7) * 16;
  const float* wptr = W + (size_t)bk * UNITS + n_blk + nloc;

  v8f acc[2][4] = {};

  const int hi = (lane >> 4) & 1;    // upper half-wave flag
  const int ml = lane & 15;

  // register staging buffers (tile in flight)
  float4 areg[4];
  float  breg[16];

  // global -> registers for K chunk at k0
  auto loadg = [&](int k0) {
#pragma unroll
    for (int i = 0; i < 4; ++i) areg[i] = *(const float4*)(xptr[i] + k0);
    const float* wp = wptr + (size_t)k0 * UNITS;
#pragma unroll
    for (int kk = 0; kk < 16; ++kk) breg[kk] = wp[(size_t)kk * UNITS];
  };

  // registers -> LDS buffer `buf` (with fused normalize + bf16 convert)
  auto stlds = [&](int buf) {
#pragma unroll
    for (int i = 0; i < 4; ++i) {
      v4bf p;
      p[0] = (__bf16)(areg[i].x * scale[i]);
      p[1] = (__bf16)(areg[i].y * scale[i]);
      p[2] = (__bf16)(areg[i].z * scale[i]);
      p[3] = (__bf16)(areg[i].w * scale[i]);
      *(v4bf*)&Alds[buf][arow[i] * LDS_STRIDE + akv[i] * 4] = p;
    }
    v8bf q0, q1;
#pragma unroll
    for (int kk = 0; kk < 8; ++kk) q0[kk] = (__bf16)breg[kk];
#pragma unroll
    for (int kk = 0; kk < 8; ++kk) q1[kk] = (__bf16)breg[kk + 8];
    *(v8bf*)&Blds[buf][nloc * LDS_STRIDE + bk]     = q0;
    *(v8bf*)&Blds[buf][nloc * LDS_STRIDE + bk + 8] = q1;
  };

  // prologue: stage tile 0
  loadg(0);
  stlds(0);
  __syncthreads();

  int cur = 0;
  for (int kt = 0; kt < D_IN / 32; ++kt) {
    const int k_next = (kt + 1) * 32;
    const bool has_next = (k_next < D_IN);
    if (has_next) loadg(k_next);  // VMEM for tile k+1 overlaps WMMA of tile k

    // --- fragment load phase -------------------------------------------
    // A frag (16-bit A 16x32): lanes 0-15: halves0-7=K0..7, 8-15=K16..23;
    //                          lanes 16-31: K8..15 and K24..31.
    v16bf af[2];
#pragma unroll
    for (int mt = 0; mt < 2; ++mt) {
      int mrow = wm * 32 + mt * 16 + ml;
      v8bf lo = *(const v8bf*)&Alds[cur][mrow * LDS_STRIDE + hi * 8];
      v8bf hh = *(const v8bf*)&Alds[cur][mrow * LDS_STRIDE + hi * 8 + 16];
#pragma unroll
      for (int i = 0; i < 8; ++i) { af[mt][i] = lo[i]; af[mt][i + 8] = hh[i]; }
    }
    // B frag (32x16): lanes 0-15 col N=lane K0..7,K16..23; lanes 16-31 K8..15,K24..31.
    v16bf bf[4];
#pragma unroll
    for (int nt = 0; nt < 4; ++nt) {
      int ncol = wn * 64 + nt * 16 + ml;
      v8bf lo = *(const v8bf*)&Blds[cur][ncol * LDS_STRIDE + hi * 8];
      v8bf hh = *(const v8bf*)&Blds[cur][ncol * LDS_STRIDE + hi * 8 + 16];
#pragma unroll
      for (int i = 0; i < 8; ++i) { bf[nt][i] = lo[i]; bf[nt][i + 8] = hh[i]; }
    }

    // Keep all 12 ds_loads before the WMMA chain: one dscnt wait, then
    // 8 back-to-back WMMAs (stlds below may interleave for co-execution).
#if __has_builtin(__builtin_amdgcn_sched_barrier)
    __builtin_amdgcn_sched_barrier(0);
#endif

    // --- compute phase: 2x4 = 8 WMMAs ----------------------------------
#pragma unroll
    for (int mt = 0; mt < 2; ++mt)
#pragma unroll
      for (int nt = 0; nt < 4; ++nt)
        acc[mt][nt] = __builtin_amdgcn_wmma_f32_16x16x32_bf16(
            false, af[mt], false, bf[nt], (short)0, acc[mt][nt], false, false);

    if (has_next) {
      stlds(cur ^ 1);   // write the OTHER buffer; readers of `cur` unaffected
      __syncthreads();  // publish tile k+1 before next iteration reads it
    }
    cur ^= 1;
  }

  // epilogue: bias + ReLU + store (C/D layout: VGPR r -> M=r (+8 upper half))
#pragma unroll
  for (int mt = 0; mt < 2; ++mt) {
    const int mo = m_blk + wm * 32 + mt * 16 + hi * 8;
#pragma unroll
    for (int nt = 0; nt < 4; ++nt) {
      int ncol = n_blk + wn * 64 + nt * 16 + ml;
      float bv = bias[ncol];
#pragma unroll
      for (int r = 0; r < 8; ++r) {
        float v = acc[mt][nt][r] + bv;
        v = v > 0.f ? v : 0.f;
        out[(size_t)(mo + r) * UNITS + ncol] = v;
      }
    }
  }
}

// ---------------------------------------------------------------------------
extern "C" void kernel_launch(void* const* d_in, const int* in_sizes, int n_in,
                              void* d_out, int out_size, void* d_ws, size_t ws_size,
                              hipStream_t stream) {
  (void)in_sizes; (void)n_in; (void)out_size; (void)ws_size;
  const float* x = (const float*)d_in[0];
  const float* W = (const float*)d_in[1];
  const float* b = (const float*)d_in[2];
  float* out      = (float*)d_out;
  float* inv_norm = (float*)d_ws;  // NROWS floats = 64 KB scratch

  rownorm_kernel<<<NROWS / 8, 256, 0, stream>>>(x, inv_norm);

  dim3 grid(UNITS / 128, NROWS / 128);  // n-major so n-blocks sharing x tiles run together
  ffdense_wmma_kernel<<<grid, 256, 0, stream>>>(x, W, b, inv_norm, out);
}
